// CircularRelativePositionAttention_30915174597079
// MI455X (gfx1250) — matrix operand order repaired and checked by reading, not compiled, not run on hardware
//
#include <hip/hip_runtime.h>
#include <hip/hip_bf16.h>

// ---------------------------------------------------------------------------
// Circular relative-position attention for MI455X (gfx1250), bf16 WMMA path.
// B=8, S=1024, D=1024, H=16, HD=64.
// One workgroup = one (b,h) + 64 query rows; 4 waves x 16-row strips (wave32).
// LDS: rel table (bf16, pair-major) 68K | T bias strips (bf16) 68K |
//      score/prob strips (bf16) 128K | 1/l (f32) 256B  => ~264 KB, 4 waves/WGP.
// ---------------------------------------------------------------------------

typedef __attribute__((ext_vector_type(16))) __bf16 v16bf;
typedef __attribute__((ext_vector_type(8)))  float  v8f;

union F16 {
  v16bf v;
  unsigned short u[16];
  unsigned int w[8];
};

// packed f32x2 -> bf16x2 (lowers to packed cvt)
__device__ __forceinline__ unsigned int pack_bf16(float x, float y) {
  __hip_bfloat162 h = __float22bfloat162_rn(make_float2(x, y));
  unsigned int r;
  __builtin_memcpy(&r, &h, sizeof(r));  // __hip_bfloat162 not trivially copyable
  return r;
}

__device__ __forceinline__ unsigned short f2bf(float f) {
  unsigned int x = __builtin_bit_cast(unsigned int, f);
  x += 0x7FFFu + ((x >> 16) & 1u);  // round-to-nearest-even
  return (unsigned short)(x >> 16);
}

__device__ __forceinline__ float bf2f(unsigned short h) {
  unsigned int u = ((unsigned int)h) << 16;  // bf16 -> f32 is exact
  return __builtin_bit_cast(float, u);
}

__device__ __forceinline__ v8f wmma_bf16(const F16 &a, const F16 &b, v8f c) {
  return __builtin_amdgcn_wmma_f32_16x16x32_bf16(false, a.v, false, b.v,
                                                 (short)0, c, false, false);
}

// A-fragment K index for 16-bit A 16x32 (contiguous order k = 16*kh + 2*vr): VGPR vr, lane-half kh.
__device__ __forceinline__ int kdmap(int vr, int kh) {
  return kh * 16 + 2 * vr;
}

constexpr int Bt = 8, Sl = 1024, Dm = 1024, Hh = 16, HD = 64;
constexpr int ND = 513;          // circular distances 0..512
constexpr int PD = 544;          // ND padded to a multiple of 32
constexpr float SCALE = 0.125f;  // 1/sqrt(64)
constexpr int NW = 4;            // waves per workgroup

// Dynamic LDS layout (bytes):
constexpr int RA_ELEMS  = PD * HD;                      // 34816 bf16
constexpr int TB_OFF    = RA_ELEMS * 2;                 // 69632
constexpr int SB_OFF    = TB_OFF + NW * 16 * PD * 2;    // 139264
constexpr int LV_OFF    = SB_OFF + NW * 16 * Sl * 2;    // 270336
constexpr int SMEM_BYTES = LV_OFF + NW * 16 * 4;        // 270592

__global__ __launch_bounds__(128)
void circ_rel_attn_kernel(const float* __restrict__ qg,
                          const float* __restrict__ kg,
                          const float* __restrict__ vg,
                          const float* __restrict__ rk,
                          const float* __restrict__ rv,
                          float* __restrict__ outg) {
  extern __shared__ char smem[];
  unsigned short* Ra  = (unsigned short*)smem;        // rel_k^T, later rel_v (bf16 pair-major)
  unsigned short* Tb  = (unsigned short*)(smem + TB_OFF);  // per-wave 16 x PD bias (bf16)
  unsigned short* Sb  = (unsigned short*)(smem + SB_OFF);  // per-wave 16 x S scores/p (bf16)
  float*          Lv  = (float*)(smem + LV_OFF);           // per-row 1/l (f32)

  const int tid  = threadIdx.x;
  const int wave = tid >> 5;
  const int lane = tid & 31;
  const int ln   = lane & 15;   // N (or M) index within fragment
  const int kh   = lane >> 4;   // lane half selects K sub-block

  const int gid  = blockIdx.x;
  const int qblk = gid & 15;          // 16 q-blocks of 64 rows
  const int h    = (gid >> 4) & 15;
  const int b    = gid >> 8;

  const int qw      = qblk * 64 + wave * 16;  // first query row of this wave
  const int headoff = h * HD;

  // ---- Phase 1: stage rel_pos_k transposed (d,j), pair-major bf16 -----------
  // pair (d0,d0+1 at column j) -> uint index (d0>>1)*PD + j
  for (int i = tid; i < RA_ELEMS / 2; i += 128) {
    const int j  = i % PD;
    const int d0 = (i / PD) * 2;
    float f0 = 0.f, f1 = 0.f;
    if (j < ND) {
      const float2 p = *(const float2*)(rk + j * HD + d0);
      f0 = p.x; f1 = p.y;
    }
    ((unsigned int*)Ra)[i] = pack_bf16(f0, f1);
  }
  __syncthreads();

  // ---- Q strip A-fragments (16 rows x 64 dims, two 32-wide K steps) ---------
  F16 aQ[2];
#pragma unroll
  for (int s = 0; s < 2; ++s) {
#pragma unroll
    for (int vr = 0; vr < 8; ++vr) {
      const int kd = s * 32 + kdmap(vr, kh);
      const float2 p =
          *(const float2*)(qg + (size_t)((b * Sl + qw + ln) * Dm + headoff + kd));
      aQ[s].w[vr] = pack_bf16(p.x, p.y);
    }
  }

  // ---- Phase 2: T[m][j] = Q[m][:] . rel_pos_k[j][:]  (WMMA, 16 x PD) --------
  for (int j0 = 0; j0 < PD; j0 += 16) {
    v8f c = {0.f, 0.f, 0.f, 0.f, 0.f, 0.f, 0.f, 0.f};
#pragma unroll
    for (int s = 0; s < 2; ++s) {
      F16 bB;
#pragma unroll
      for (int vr = 0; vr < 8; ++vr) {
        const int kd0 = s * 32 + kh * 16 + 2 * vr;  // even K row pair
        bB.w[vr] = *(const unsigned int*)(Ra + (((kd0 >> 1) * PD + j0 + ln) << 1));
      }
      c = wmma_bf16(aQ[s], bB, c);
    }
#pragma unroll
    for (int vr = 0; vr < 8; ++vr)
      Tb[(wave * 16 + vr + 8 * kh) * PD + j0 + ln] = f2bf(c[vr]);
  }

  // ---- Phase 3: score strip = QK^T * scale + T[mind]  -> Sb (bf16) ----------
  for (int j0 = 0; j0 < Sl; j0 += 16) {
    if (j0 + 16 < Sl)
      __builtin_prefetch(
          kg + (size_t)((b * Sl + j0 + 16 + ln) * Dm + headoff + kh * 32), 0, 0);
    v8f c = {0.f, 0.f, 0.f, 0.f, 0.f, 0.f, 0.f, 0.f};
#pragma unroll
    for (int s = 0; s < 2; ++s) {
      F16 bB;  // B[kd][n] = K[j0+n][kd]
#pragma unroll
      for (int vr = 0; vr < 8; ++vr) {
        const int kd0 = s * 32 + kh * 16 + 2 * vr;
        const float2 p =
            *(const float2*)(kg + (size_t)((b * Sl + j0 + ln) * Dm + headoff + kd0));
        bB.w[vr] = pack_bf16(p.x, p.y);
      }
      c = wmma_bf16(aQ[s], bB, c);
    }
#pragma unroll
    for (int vr = 0; vr < 8; ++vr) {
      const int m    = vr + 8 * kh;
      const int qrow = qw + m;
      const int kcol = j0 + ln;
      const int dd   = (qrow - kcol) & (Sl - 1);
      const int mind = dd < Sl - dd ? dd : Sl - dd;
      const float bias = bf2f(Tb[(wave * 16 + m) * PD + mind]);
      Sb[(wave * 16 + m) * Sl + kcol] = f2bf(c[vr] * SCALE + bias);
    }
  }

  // ---- Softmax per row (2 lanes/row), keep un-normalized p, stash 1/l -------
  {
    const int row = lane >> 1, half = lane & 1;
    uint4* r4 = (uint4*)(Sb + (wave * 16 + row) * Sl + half * 512);  // 64 x 8 bf16
    float mx = -3.4e38f;
    for (int i = 0; i < 64; ++i) {
      const uint4 u = r4[i];
      const unsigned int ws[4] = {u.x, u.y, u.z, u.w};
#pragma unroll
      for (int t = 0; t < 4; ++t) {
        mx = fmaxf(mx, fmaxf(bf2f((unsigned short)(ws[t] & 0xffffu)),
                             bf2f((unsigned short)(ws[t] >> 16))));
      }
    }
    mx = fmaxf(mx, __shfl_xor(mx, 1, 32));
    float sum = 0.f;
    for (int i = 0; i < 64; ++i) {
      uint4 u = r4[i];
      unsigned int ws[4] = {u.x, u.y, u.z, u.w};
#pragma unroll
      for (int t = 0; t < 4; ++t) {
        const float e0 = __expf(bf2f((unsigned short)(ws[t] & 0xffffu)) - mx);
        const float e1 = __expf(bf2f((unsigned short)(ws[t] >> 16)) - mx);
        sum += e0 + e1;
        ws[t] = pack_bf16(e0, e1);
      }
      u.x = ws[0]; u.y = ws[1]; u.z = ws[2]; u.w = ws[3];
      r4[i] = u;
    }
    sum += __shfl_xor(sum, 1, 32);
    if (half == 0) Lv[wave * 16 + row] = 1.f / sum;
  }
  __syncthreads();  // all waves done with Ra (phase 2) and own strips

  // ---- Phase 4: overwrite Ra with rel_pos_v (row=kdist, pair-major bf16) ----
  // pair (kdist0, kdist0+1 at dim d) -> uint index (kdist0>>1)*HD + d
  for (int i = tid; i < RA_ELEMS / 2; i += 128) {
    const int d  = i & 63;
    const int k0 = (i >> 6) * 2;
    const float f0 = (k0 < ND) ? rv[k0 * HD + d] : 0.f;
    const float f1 = (k0 + 1 < ND) ? rv[(k0 + 1) * HD + d] : 0.f;
    ((unsigned int*)Ra)[i] = pack_bf16(f0, f1);
  }
  __syncthreads();

  // ---- Phase 5: O = P @ V + W @ rel_v, then scale by 1/l --------------------
  v8f acc[4];
#pragma unroll
  for (int t = 0; t < 4; ++t)
    acc[t] = (v8f){0.f, 0.f, 0.f, 0.f, 0.f, 0.f, 0.f, 0.f};

  // P @ V over all 1024 keys in 32-wide K steps (A: packed bf16 p, no cvt)
  for (int kb = 0; kb < Sl; kb += 32) {
    F16 aP;
#pragma unroll
    for (int vr = 0; vr < 8; ++vr) {
      const int kd = kdmap(vr, kh);
      aP.w[vr] = *(const unsigned int*)(Sb + (wave * 16 + ln) * Sl + kb + kd);
    }
    if (kb + 32 < Sl)
      __builtin_prefetch(
          vg + (size_t)((b * Sl + kb + 32 + kh * 16) * Dm + headoff + ln), 0, 0);
#pragma unroll
    for (int t = 0; t < 4; ++t) {
      const int dcol = t * 16 + ln;
      F16 bB;  // B[k][d] = V[k][d]
#pragma unroll
      for (int vr = 0; vr < 8; ++vr) {
        const int k0 = kb + kh * 16 + 2 * vr;
        const float f0 = vg[(size_t)((b * Sl + k0) * Dm + headoff + dcol)];
        const float f1 = vg[(size_t)((b * Sl + k0 + 1) * Dm + headoff + dcol)];
        bB.w[vr] = pack_bf16(f0, f1);
      }
      acc[t] = wmma_bf16(aP, bB, acc[t]);
    }
  }

  // W @ rel_v ; W[q][d] = p[q][(q-d)%S] (+ p[q][(q+d)%S] when distinct), d<=512
  {
    const int qrow = qw + ln;
    const unsigned short* prow = Sb + (wave * 16 + ln) * Sl;
    for (int kb = 0; kb < PD; kb += 32) {
      F16 aW;
#pragma unroll
      for (int vr = 0; vr < 8; ++vr) {
        const int kd = kb + kdmap(vr, kh);
        float wv[2];
#pragma unroll
        for (int e = 0; e < 2; ++e) {
          const int d  = kd + e;
          const int k1 = (qrow - d) & (Sl - 1);
          const int k2 = (qrow + d) & (Sl - 1);
          float w = bf2f(prow[k1]) + ((k2 != k1) ? bf2f(prow[k2]) : 0.f);
          wv[e] = (d <= 512) ? w : 0.f;
        }
        aW.w[vr] = pack_bf16(wv[0], wv[1]);
      }
#pragma unroll
      for (int t = 0; t < 4; ++t) {
        const int dcol = t * 16 + ln;
        F16 bB;  // B[kdist][d] = rel_v[kdist][d], pair-major in LDS
#pragma unroll
        for (int vr = 0; vr < 8; ++vr) {
          const int kd0 = kb + kh * 16 + 2 * vr;
          bB.w[vr] = *(const unsigned int*)(Ra + (((kd0 >> 1) * HD + dcol) << 1));
        }
        acc[t] = wmma_bf16(aW, bB, acc[t]);
      }
    }
  }

  // Final store with 1/l normalization (C layout: lane=N, rows vr + 8*kh)
#pragma unroll
  for (int vr = 0; vr < 8; ++vr) {
    const int m = vr + 8 * kh;
    const float linv = Lv[wave * 16 + m];
    const int qrow = qw + m;
#pragma unroll
    for (int t = 0; t < 4; ++t) {
      outg[(size_t)((b * Sl + qrow) * Dm + headoff + t * 16 + ln)] =
          acc[t][vr] * linv;
    }
  }
}

extern "C" void kernel_launch(void* const* d_in, const int* in_sizes, int n_in,
                              void* d_out, int out_size, void* d_ws, size_t ws_size,
                              hipStream_t stream) {
  (void)in_sizes; (void)n_in; (void)out_size; (void)d_ws; (void)ws_size;
  const float* q  = (const float*)d_in[0];
  const float* k  = (const float*)d_in[1];
  const float* v  = (const float*)d_in[2];
  const float* rk = (const float*)d_in[3];
  const float* rv = (const float*)d_in[4];
  float* out = (float*)d_out;

  // Idempotent + deterministic; safe to call every invocation.
  (void)hipFuncSetAttribute((const void*)circ_rel_attn_kernel,
                            hipFuncAttributeMaxDynamicSharedMemorySize,
                            SMEM_BYTES);

  // grid: B * H * (S/64) = 8 * 16 * 16 = 2048 workgroups, 4 waves each
  circ_rel_attn_kernel<<<dim3(Bt * Hh * (Sl / 64)), dim3(128), SMEM_BYTES, stream>>>(
      q, k, v, rk, rv, out);
}
